// NetVLAD_35639638622780
// MI455X (gfx1250) — matrix operand order repaired and checked by reading, not compiled, not run on hardware
//
#include <hip/hip_runtime.h>
#include <hip/hip_bf16.h>
#include <math.h>

typedef __attribute__((ext_vector_type(16))) _Float16 v16h;
typedef __attribute__((ext_vector_type(8)))  float    v8f;

#define N_IMG   32
#define C_DIM   128
#define L_DIM   4096
#define K_CL    64
#define LB      128
#define NTILES  (L_DIM / LB)
#define XPAD    4
#define HPAD    16
#define APAD    8

__global__ __launch_bounds__(256, 1)
void netvlad_fused_kernel(const float* __restrict__ x,
                          const float* __restrict__ conv_w,
                          const float* __restrict__ conv_b,
                          const float* __restrict__ centroids,
                          float* __restrict__ out)
{
  // f32 staging tile; later overlaid by logits[K][LB] and final vlad[K][C]
  __shared__ alignas(32) float     s_xf[C_DIM][LB + XPAD];     // 66 KB
  __shared__ alignas(32) _Float16  s_xh[C_DIM][LB + HPAD];     // 36 KB  (normalized f16)
  __shared__ alignas(32) _Float16  s_wfrag[4][4][32][16];      // 16 KB  (W pre-swizzled A-frags)
  __shared__ alignas(32) _Float16  s_a[K_CL][LB + APAD];       // 17 KB  (softmax result)
  __shared__ float s_rnorm[LB];
  __shared__ float s_bias[K_CL];
  __shared__ float s_asum[K_CL];
  __shared__ float s_rowinv[K_CL];
  __shared__ float s_tot;

  const int n    = blockIdx.x;
  const int tid  = threadIdx.x;          // 256 threads = 8 wave32s
  const int wave = tid >> 5;
  const int lane = tid & 31;
  const int mt   = wave & 3;             // cluster-row tile (rows 16*mt..16*mt+15)
  const int grp  = wave >> 2;            // 0/1 : which half of the N dimension
  const int lgrp = lane >> 4;            // 0/1 : lane half-group
  const int lm   = lane & 15;

  // ---- one-time staging: bias + W swizzled into per-lane A fragments ----
  if (tid < K_CL) s_bias[tid] = conv_b[tid];
  for (int i = tid; i < 4 * 4 * 32 * 16; i += 256) {
    const int h   = i & 15;
    const int ln  = (i >> 4) & 31;
    const int ks  = (i >> 9) & 3;
    const int m_t = i >> 11;
    // CDNA5 16-bit A-matrix 16x32 layout: lane half-group offsets k by 8,
    // VGPRs 0..3 hold k 0..7(+8), VGPRs 4..7 hold k 16..23(+8)
    const int v  = h >> 1;
    const int kk = ((v < 4) ? (2 * v + (h & 1)) : (16 + 2 * (v - 4) + (h & 1)))
                   + ((ln >> 4) << 3);
    const int m  = (m_t << 4) + (ln & 15);
    s_wfrag[m_t][ks][ln][h] = (_Float16)conv_w[m * C_DIM + (ks << 5) + kk];
  }
  __syncthreads();

  float asum_reg = 0.0f;                 // thread t<64 owns cluster t
  v8f vacc[4];                           // whole vlad[64][128] lives in registers
#pragma unroll
  for (int j = 0; j < 4; ++j)
#pragma unroll
    for (int r = 0; r < 8; ++r) vacc[j][r] = 0.0f;

  const float* __restrict__ xn = x + (size_t)n * C_DIM * L_DIM;

  for (int t = 0; t < NTILES; ++t) {
    const int l0   = t * LB;
    const int lcol = tid & (LB - 1);
    const int crow = tid >> 7;

    // ---- Phase 1: coalesced global -> LDS (f32), prefetch next tile to L2 ----
    for (int c = crow; c < C_DIM; c += 2) {
      s_xf[c][lcol] = xn[(size_t)c * L_DIM + l0 + lcol];
      if (t + 1 < NTILES)
        __builtin_prefetch(&xn[(size_t)c * L_DIM + l0 + LB + lcol], 0, 1);
    }
    __syncthreads();

    // ---- Phase 2: per-location channel L2 norm ----
    if (tid < LB) {
      float ss = 0.0f;
      for (int c = 0; c < C_DIM; ++c) { float v = s_xf[c][tid]; ss += v * v; }
      s_rnorm[tid] = 1.0f / fmaxf(sqrtf(ss), 1e-12f);
    }
    __syncthreads();

    // ---- Phase 3: normalize + convert to f16 ----
    {
      const float rn = s_rnorm[lcol];
      for (int c = crow; c < C_DIM; c += 2)
        s_xh[c][lcol] = (_Float16)(s_xf[c][lcol] * rn);
    }
    __syncthreads();

    // ---- Phase 4: GEMM1 logits[K,LB] = W * xhat + b  (WMMA f16->f32) ----
    float (*s_logit)[LB + XPAD] = s_xf;  // overlay: raw f32 tile is dead now
    {
      v16h afr[4];
#pragma unroll
      for (int ks = 0; ks < 4; ++ks)
        afr[ks] = *(const v16h*)&s_wfrag[mt][ks][lane][0];  // contiguous 32B/lane
#pragma unroll
      for (int q = 0; q < 4; ++q) {
        const int lt   = (grp << 2) + q;
        const int lidx = (lt << 4) + lm;
        v8f acc;
#pragma unroll
        for (int r = 0; r < 8; ++r)
          acc[r] = s_bias[(mt << 4) + (lgrp << 3) + r];     // bias preloaded into C
#pragma unroll
        for (int ks = 0; ks < 4; ++ks) {
          v16h bfr;                                          // B: k=c in split-8 order, same as A
#pragma unroll
          for (int h = 0; h < 16; ++h)
            bfr[h] = s_xh[(ks << 5) + (lgrp << 3) + (h & 7) + ((h >> 3) << 4)][lidx];
          acc = __builtin_amdgcn_wmma_f32_16x16x32_f16(false, afr[ks], false, bfr,
                                                       (short)0, acc, false, false);
        }
#pragma unroll
        for (int r = 0; r < 8; ++r)
          s_logit[(mt << 4) + (lgrp << 3) + r][lidx] = acc[r];
      }
    }
    __syncthreads();

    // ---- Phase 5: softmax over 64 clusters per location ----
    if (tid < LB) {
      float mx = -3.4e38f;
      for (int k = 0; k < K_CL; ++k) mx = fmaxf(mx, s_logit[k][tid]);
      float s = 0.0f;
      for (int k = 0; k < K_CL; ++k) s += __expf(s_logit[k][tid] - mx);
      const float rs = 1.0f / s;
      for (int k = 0; k < K_CL; ++k)
        s_a[k][tid] = (_Float16)(__expf(s_logit[k][tid] - mx) * rs);
    }
    __syncthreads();

    if (tid < K_CL) {                    // asum[k] += sum_l a[k,l]
      float s = 0.0f;
      for (int l = 0; l < LB; ++l) s += (float)s_a[tid][l];
      asum_reg += s;
    }

    // ---- Phase 6: GEMM2 vlad += a[K,LB] * xhat^T[LB,C]  (WMMA, reg-resident C) ----
    {
#pragma unroll
      for (int ks = 0; ks < 4; ++ks) {
        v16h afr;
        const int m = (mt << 4) + lm;
#pragma unroll
        for (int h = 0; h < 16; ++h) {
          const int kk = (lgrp << 4) + h;
          afr[h] = s_a[m][(ks << 5) + kk];
        }
        const int lbase = (ks << 5) + (lgrp << 4);
#pragma unroll
        for (int j = 0; j < 4; ++j) {
          const int cc = (((grp << 2) + j) << 4) + lm;
          v16h bfr = *(const v16h*)&s_xh[cc][lbase];          // contiguous 32B/lane
          vacc[j] = __builtin_amdgcn_wmma_f32_16x16x32_f16(false, afr, false, bfr,
                                                           (short)0, vacc[j], false, false);
        }
      }
    }
    __syncthreads();  // protect s_xh / s_a / s_xf before next tile
  }

  // ---- Endgame: vlad -= asum*centroid, intra-norm, global norm, store ----
  if (tid < K_CL) s_asum[tid] = asum_reg;
  __syncthreads();

  float (*s_vlad)[LB + XPAD] = s_xf;     // overlay again
#pragma unroll
  for (int j = 0; j < 4; ++j)
#pragma unroll
    for (int r = 0; r < 8; ++r) {
      const int m = (mt << 4) + (lgrp << 3) + r;
      const int c = (((grp << 2) + j) << 4) + lm;
      s_vlad[m][c] = vacc[j][r] - s_asum[m] * centroids[m * C_DIM + c];
    }
  if (tid == 0) s_tot = 0.0f;
  __syncthreads();

  if (tid < K_CL) {
    float ss = 0.0f;
    for (int c = 0; c < C_DIM; ++c) { float v = s_vlad[tid][c]; ss += v * v; }
    const float ri = 1.0f / fmaxf(sqrtf(ss), 1e-12f);
    s_rowinv[tid] = ri;
    atomicAdd(&s_tot, ss * ri * ri);
  }
  __syncthreads();

  const float rtot = 1.0f / fmaxf(sqrtf(s_tot), 1e-12f);
  float* outn = out + (size_t)n * K_CL * C_DIM;
  for (int i = tid; i < K_CL * C_DIM; i += 256) {
    const int k = i >> 7;
    const int c = i & 127;
    outn[i] = s_vlad[k][c] * s_rowinv[k] * rtot;
  }
}

extern "C" void kernel_launch(void* const* d_in, const int* in_sizes, int n_in,
                              void* d_out, int out_size, void* d_ws, size_t ws_size,
                              hipStream_t stream) {
  const float* x         = (const float*)d_in[0];
  const float* conv_w    = (const float*)d_in[1];
  const float* conv_b    = (const float*)d_in[2];
  const float* centroids = (const float*)d_in[3];
  float* out = (float*)d_out;
  (void)in_sizes; (void)n_in; (void)d_ws; (void)ws_size; (void)out_size;
  netvlad_fused_kernel<<<dim3(N_IMG), dim3(256), 0, stream>>>(x, conv_w, conv_b,
                                                              centroids, out);
}
